// TopKSAE_128849019036
// MI455X (gfx1250) — matrix operand order repaired and checked by reading, compiled, not run on hardware
//
#include <hip/hip_runtime.h>

#define D_MODEL 2048
#define D_SAE   32768
#define N_TOK   4096
#define TOPK    64

typedef __attribute__((ext_vector_type(16))) __bf16       v16bf;
typedef __attribute__((ext_vector_type(8)))  float        v8f;
typedef __attribute__((ext_vector_type(2)))  unsigned int u32x2;
typedef __attribute__((ext_vector_type(4)))  unsigned int u32x4;
typedef __attribute__((ext_vector_type(8)))  unsigned int u32x8;

// Pack bf16(a) | bf16(b)<<16 via one v_perm_b32 (truncation rounding).
__device__ __forceinline__ unsigned int f2bf2(float a, float b) {
  const unsigned int ua = __float_as_uint(a);
  const unsigned int ub = __float_as_uint(b);
#if __has_builtin(__builtin_amdgcn_perm)
  // concat {hi=ub, lo=ua}; select bytes {ub[3],ub[2],ua[3],ua[2]}
  return __builtin_amdgcn_perm(ub, ua, 0x07060302u);
#else
  return (ua >> 16) | (ub & 0xFFFF0000u);
#endif
}

// Truncated bf16 of a float's top 16 bits (pattern-matches ds_store_b16_d16_hi)
__device__ __forceinline__ unsigned short f2bf_hi(float a) {
  return (unsigned short)(__float_as_uint(a) >> 16);
}

// ---------------------------------------------------------------------------
// Phase 1: pre = x @ W_enc + b_enc   (bf16 WMMA, f32 accumulate)
// Workgroup tile 128x128, BK=32, 256 threads = 8 waves (4 M x 2 N),
// wave tile 32x64 = 2x4 v8f accumulators -> 8 v_wmma_f32_16x16x32_bf16 / step.
// ---------------------------------------------------------------------------
#define BM  128
#define BN  128
#define BK  32
#define LDT 40   // ushort stride per LDS tile row (32 data + 8 pad, 80B)

__global__ __launch_bounds__(256)
void sae_encode_wmma(const float* __restrict__ x, const float* __restrict__ W,
                     const float* __restrict__ b_enc, float* __restrict__ pre) {
  __shared__ unsigned short As[BM * LDT];  // [m][k], bf16
  __shared__ unsigned short Bs[BN * LDT];  // [n][k], bf16 (transposed at store)

  const int tid  = threadIdx.x;
  const int lane = tid & 31;
  const int wv   = tid >> 5;
  const int wm   = wv & 3;        // 4 waves along M  (4 * 32 = 128)
  const int wn   = wv >> 2;       // 2 waves along N  (2 * 64 = 128)
  const int hf   = lane >> 4;     // lane half (0/1)
  const int lm   = lane & 15;

  const int blockN = blockIdx.x * BN;
  const int blockM = blockIdx.y * BM;

  const v8f vzero = {0.f, 0.f, 0.f, 0.f, 0.f, 0.f, 0.f, 0.f};
  v8f acc[2][4];
#pragma unroll
  for (int i = 0; i < 2; ++i)
#pragma unroll
    for (int j = 0; j < 4; ++j) acc[i][j] = vzero;

  float4 a_pf[4], b_pf[4];

  // ---- prologue: load K-tile 0 into registers, stage to LDS ----
#pragma unroll
  for (int i = 0; i < 4; ++i) {
    const int f  = i * 256 + tid;           // 0..1023
    const int m  = f >> 3, kq = f & 7;      // A: 128 rows x 8 float4
    a_pf[i] = *(const float4*)(x + (size_t)(blockM + m) * D_MODEL + kq * 4);
    const int kk = f >> 5, nq = f & 31;     // B: 32 rows x 32 float4
    b_pf[i] = *(const float4*)(W + (size_t)kk * D_SAE + blockN + nq * 4);
  }
#pragma unroll
  for (int i = 0; i < 4; ++i) {
    const int f  = i * 256 + tid;
    const int m  = f >> 3, kq = f & 7;
    u32x2 p;
    p.x = f2bf2(a_pf[i].x, a_pf[i].y);
    p.y = f2bf2(a_pf[i].z, a_pf[i].w);
    *(u32x2*)(As + m * LDT + kq * 4) = p;
    const int kk = f >> 5, n0 = (f & 31) * 4;
    Bs[(n0 + 0) * LDT + kk] = f2bf_hi(b_pf[i].x);
    Bs[(n0 + 1) * LDT + kk] = f2bf_hi(b_pf[i].y);
    Bs[(n0 + 2) * LDT + kk] = f2bf_hi(b_pf[i].z);
    Bs[(n0 + 3) * LDT + kk] = f2bf_hi(b_pf[i].w);
  }
  __syncthreads();

  for (int kt = 0; kt < D_MODEL / BK; ++kt) {
    // ---- prefetch next K-tile into registers (overlaps with WMMA) ----
    if (kt < D_MODEL / BK - 1) {
      const int k0 = (kt + 1) * BK;
#pragma unroll
      for (int i = 0; i < 4; ++i) {
        const int f  = i * 256 + tid;
        const int m  = f >> 3, kq = f & 7;
        a_pf[i] = *(const float4*)(x + (size_t)(blockM + m) * D_MODEL + k0 + kq * 4);
        const int kk = f >> 5, nq = f & 31;
        b_pf[i] = *(const float4*)(W + (size_t)(k0 + kk) * D_SAE + blockN + nq * 4);
      }
    }

    // ---- load fragments from LDS (ISA-native layouts, 2x ds_load_b128 each)
    v16bf afrag[2], bfrag[4];
#pragma unroll
    for (int i = 0; i < 2; ++i) {
      const unsigned short* base = As + (wm * 32 + i * 16 + lm) * LDT;
      u32x4 lo = *(const u32x4*)(base + 8 * hf);        // K = 8h .. 8h+7
      u32x4 hi = *(const u32x4*)(base + 16 + 8 * hf);   // K = 16+8h .. 16+8h+7
      u32x8 t;
      t[0] = lo[0]; t[1] = lo[1]; t[2] = lo[2]; t[3] = lo[3];
      t[4] = hi[0]; t[5] = hi[1]; t[6] = hi[2]; t[7] = hi[3];
      afrag[i] = __builtin_bit_cast(v16bf, t);
    }
#pragma unroll
    for (int j = 0; j < 4; ++j) {
      const unsigned short* base = Bs + (wn * 64 + j * 16 + lm) * LDT;
      u32x4 lo = *(const u32x4*)(base + 8 * hf);        // K = 8h .. 8h+7
      u32x4 hi = *(const u32x4*)(base + 16 + 8 * hf);   // K = 16+8h .. 16+8h+7
      u32x8 t;
      t[0] = lo[0]; t[1] = lo[1]; t[2] = lo[2]; t[3] = lo[3];
      t[4] = hi[0]; t[5] = hi[1]; t[6] = hi[2]; t[7] = hi[3];
      bfrag[j] = __builtin_bit_cast(v16bf, t);
    }

    // ---- 8 WMMAs ----
#pragma unroll
    for (int i = 0; i < 2; ++i)
#pragma unroll
      for (int j = 0; j < 4; ++j)
        acc[i][j] = __builtin_amdgcn_wmma_f32_16x16x32_bf16(
            false, afrag[i], false, bfrag[j], (short)0, acc[i][j], false, false);

    __syncthreads();
    if (kt < D_MODEL / BK - 1) {
#pragma unroll
      for (int i = 0; i < 4; ++i) {
        const int f  = i * 256 + tid;
        const int m  = f >> 3, kq = f & 7;
        u32x2 p;
        p.x = f2bf2(a_pf[i].x, a_pf[i].y);
        p.y = f2bf2(a_pf[i].z, a_pf[i].w);
        *(u32x2*)(As + m * LDT + kq * 4) = p;
        const int kk = f >> 5, n0 = (f & 31) * 4;
        Bs[(n0 + 0) * LDT + kk] = f2bf_hi(b_pf[i].x);
        Bs[(n0 + 1) * LDT + kk] = f2bf_hi(b_pf[i].y);
        Bs[(n0 + 2) * LDT + kk] = f2bf_hi(b_pf[i].z);
        Bs[(n0 + 3) * LDT + kk] = f2bf_hi(b_pf[i].w);
      }
    }
    __syncthreads();
  }

  // ---- epilogue: add b_enc, write pre (C layout: VGPR r -> M=r+8h, N=lm) ----
#pragma unroll
  for (int j = 0; j < 4; ++j) {
    const int col  = blockN + wn * 64 + j * 16 + lm;
    const float bias = b_enc[col];
#pragma unroll
    for (int i = 0; i < 2; ++i) {
#pragma unroll
      for (int r = 0; r < 8; ++r) {
        const int rowg = blockM + wm * 32 + i * 16 + 8 * hf + r;
        pre[(size_t)rowg * D_SAE + col] = acc[i][j][r] + bias;
      }
    }
  }
}

// ---------------------------------------------------------------------------
// Phase 2: per-token top-64 of relu(pre) via 4-pass byte radix select,
// zero-fill + scatter z, emit compact (idx,val) for decode.
// ---------------------------------------------------------------------------
__global__ __launch_bounds__(256)
void sae_topk_scatter(const float* __restrict__ pre, float* __restrict__ z,
                      int* __restrict__ ws_idx, float* __restrict__ ws_val) {
  const int token = blockIdx.x;
  const int tid   = threadIdx.x;
  const float* row  = pre + (size_t)token * D_SAE;
  float*       zrow = z   + (size_t)token * D_SAE;

  __shared__ unsigned int hist[256];
  __shared__ unsigned int s_prefix;
  __shared__ int          s_remaining;
  __shared__ unsigned int s_tie, s_cnt;

  unsigned int prefix = 0;
  int remaining = TOPK;

  // radix select the 64th-largest key (non-negative floats are order-isomorphic to uint)
  for (int level = 3; level >= 0; --level) {
    hist[tid] = 0;
    __syncthreads();
    const unsigned int mask = (level == 3) ? 0u : (0xFFFFFFFFu << ((level + 1) * 8));
    for (int i = tid; i < D_SAE; i += 256) {
      const float v = row[i];
      const unsigned int key = (v > 0.0f) ? __float_as_uint(v) : 0u;
      if ((key & mask) == prefix)
        atomicAdd(&hist[(key >> (level * 8)) & 0xFF], 1u);
    }
    __syncthreads();
    if (tid == 0) {
      unsigned int cum = 0;
      int b;
      for (b = 255; b > 0; --b) {
        const unsigned int h = hist[b];
        if (cum + h >= (unsigned int)remaining) break;
        cum += h;
      }
      s_prefix    = prefix | ((unsigned int)b << (level * 8));
      s_remaining = remaining - (int)cum;
    }
    __syncthreads();
    prefix    = s_prefix;
    remaining = s_remaining;
  }
  const unsigned int threshold = prefix;

  if (tid == 0) { s_tie = 0; s_cnt = 0; }
  if (tid < TOPK) {  // safe defaults in case of <K positives
    ws_idx[(size_t)token * TOPK + tid] = 0;
    ws_val[(size_t)token * TOPK + tid] = 0.0f;
  }
  // zero-fill z row (d_out is poisoned)
  const float4 z4 = make_float4(0.f, 0.f, 0.f, 0.f);
  for (int i = tid; i < D_SAE / 4; i += 256) ((float4*)zrow)[i] = z4;
  __syncthreads();

  // exact-K selection with tie cap
  for (int i = tid; i < D_SAE; i += 256) {
    const float v = row[i];
    const unsigned int key = (v > 0.0f) ? __float_as_uint(v) : 0u;
    bool sel = key > threshold;
    if (!sel && key == threshold) {
      const unsigned int p = atomicAdd(&s_tie, 1u);
      sel = (p < (unsigned int)remaining);
    }
    if (sel) {
      const unsigned int slot = atomicAdd(&s_cnt, 1u);
      const float val = (v > 0.0f) ? v : 0.0f;
      if (slot < TOPK) {
        ws_idx[(size_t)token * TOPK + slot] = i;
        ws_val[(size_t)token * TOPK + slot] = val;
        zrow[i] = val;
      }
    }
  }
}

// ---------------------------------------------------------------------------
// Phase 3: x_hat = z @ W_dec + b_dec via sparse gather (64 rows per token)
// ---------------------------------------------------------------------------
__global__ __launch_bounds__(256)
void sae_decode(const float* __restrict__ W_dec, const float* __restrict__ b_dec,
                const int* __restrict__ ws_idx, const float* __restrict__ ws_val,
                float* __restrict__ xhat) {
  const int token = blockIdx.x;
  const int tid   = threadIdx.x;
  __shared__ int   s_idx[TOPK];
  __shared__ float s_val[TOPK];
  if (tid < TOPK) {
    s_idx[tid] = ws_idx[(size_t)token * TOPK + tid];
    s_val[tid] = ws_val[(size_t)token * TOPK + tid];
  }
  __syncthreads();

  float4 a0 = ((const float4*)b_dec)[tid * 2 + 0];
  float4 a1 = ((const float4*)b_dec)[tid * 2 + 1];
#pragma unroll 8
  for (int k = 0; k < TOPK; ++k) {
    const float v = s_val[k];
    const float4* wr = (const float4*)(W_dec + (size_t)s_idx[k] * D_MODEL);
    const float4 w0 = wr[tid * 2 + 0];
    const float4 w1 = wr[tid * 2 + 1];
    a0.x = fmaf(v, w0.x, a0.x); a0.y = fmaf(v, w0.y, a0.y);
    a0.z = fmaf(v, w0.z, a0.z); a0.w = fmaf(v, w0.w, a0.w);
    a1.x = fmaf(v, w1.x, a1.x); a1.y = fmaf(v, w1.y, a1.y);
    a1.z = fmaf(v, w1.z, a1.z); a1.w = fmaf(v, w1.w, a1.w);
  }
  float4* out = (float4*)(xhat + (size_t)token * D_MODEL);
  out[tid * 2 + 0] = a0;
  out[tid * 2 + 1] = a1;
}

// ---------------------------------------------------------------------------
extern "C" void kernel_launch(void* const* d_in, const int* in_sizes, int n_in,
                              void* d_out, int out_size, void* d_ws, size_t ws_size,
                              hipStream_t stream) {
  const float* x     = (const float*)d_in[0];
  const float* W_enc = (const float*)d_in[1];
  const float* b_enc = (const float*)d_in[2];
  const float* W_dec = (const float*)d_in[3];
  const float* b_dec = (const float*)d_in[4];

  // outputs concatenated in return order: x_hat, z, pre
  float* out_xhat = (float*)d_out;
  float* out_z    = out_xhat + (size_t)N_TOK * D_MODEL;
  float* out_pre  = out_z    + (size_t)N_TOK * D_SAE;

  int*   ws_idx = (int*)d_ws;
  float* ws_val = (float*)((char*)d_ws + (size_t)N_TOK * TOPK * sizeof(int));

  dim3 ggrid(D_SAE / BN, N_TOK / BM);  // 256 x 32 workgroups
  sae_encode_wmma<<<ggrid, dim3(256), 0, stream>>>(x, W_enc, b_enc, out_pre);
  sae_topk_scatter<<<dim3(N_TOK), dim3(256), 0, stream>>>(out_pre, out_z, ws_idx, ws_val);
  sae_decode<<<dim3(N_TOK), dim3(256), 0, stream>>>(W_dec, b_dec, ws_idx, ws_val, out_xhat);

  (void)in_sizes; (void)n_in; (void)out_size; (void)ws_size;
}
